// Head_72885595013521
// MI455X (gfx1250) — matrix operand order repaired and checked by reading, not compiled, not run on hardware
//
#include <hip/hip_runtime.h>
#include <math.h>

#define BB 4
#define TT 4096
#define CC 1024
#define HH 64
#define NT (TT/16)   // 256 query tiles per batch

typedef __attribute__((ext_vector_type(16))) _Float16 v16h;
typedef __attribute__((ext_vector_type(8)))  _Float16 v8h;
typedef __attribute__((ext_vector_type(4)))  _Float16 v4h;
typedef __attribute__((ext_vector_type(8)))  float    v8f;
typedef __attribute__((ext_vector_type(4)))  float    v4f;

// ---------------------------------------------------------------------------
// Kernel 0: one-shot f32 -> f16 conversion of the three weight matrices.
// 3 x 128 KB f16, stays L2-resident for the projection kernel.
// ---------------------------------------------------------------------------
__global__ __launch_bounds__(256) void wcvt_kernel(
    const float* __restrict__ Wq, const float* __restrict__ Wk,
    const float* __restrict__ Wv,
    _Float16* __restrict__ Whq, _Float16* __restrict__ Whk,
    _Float16* __restrict__ Whv)
{
    const int i = (blockIdx.x * 256 + threadIdx.x) * 4;
    v4f q = *(const v4f*)(Wq + i);
    v4f k = *(const v4f*)(Wk + i);
    v4f v = *(const v4f*)(Wv + i);
    v4h qh, kh, vh;
    #pragma unroll
    for (int e = 0; e < 4; ++e) {
        qh[e] = (_Float16)q[e];
        kh[e] = (_Float16)k[e];
        vh[e] = (_Float16)v[e];
    }
    *(v4h*)(Whq + i) = qh;
    *(v4h*)(Whk + i) = kh;
    *(v4h*)(Whv + i) = vh;
}

// B-fragment (32x16 f16) of W^T: column h of W^T = row h of Wh (contiguous).
// Lanes 0-15 hold K {0..7,16..23} (N = lane), lanes 16-31 hold K {8..15,24..31}.
static __device__ inline v16h load_w_bfrag(const _Float16* __restrict__ Wh,
                                           int n, int lr, int hf, int kk)
{
    const _Float16* wp = Wh + (size_t)(n*16 + lr)*CC + kk*32 + hf*8;
    v8h w0 = *(const v8h*)(wp);
    v8h w1 = *(const v8h*)(wp + 16);
    v16h r;
    #pragma unroll
    for (int e = 0; e < 8; ++e) { r[e] = w0[e]; r[8+e] = w1[e]; }
    return r;
}

// ---------------------------------------------------------------------------
// Kernel 1: QKV projection (WMMA f16->f32) + fused RoPE, writes f16 Q/K ([t][h])
// and transposed V ([h][t]) into workspace. One wave per 16-row tile.
// ---------------------------------------------------------------------------
__global__ __launch_bounds__(128) void qkv_rope_kernel(
    const float* __restrict__ x,
    const _Float16* __restrict__ Whq, const _Float16* __restrict__ Whk,
    const _Float16* __restrict__ Whv,
    _Float16* __restrict__ Qh, _Float16* __restrict__ Kh,
    _Float16* __restrict__ Vt)
{
    const int lane = threadIdx.x & 31;
    const int wave = threadIdx.x >> 5;
    const int tile = blockIdx.x * 4 + wave;          // 0 .. B*NT-1
    const int b  = tile / NT;
    const int t0 = (tile % NT) * 16;
    const int hf = lane >> 4;                        // half-wave id
    const int lr = lane & 15;

    const v8f vzero = {0.f,0.f,0.f,0.f,0.f,0.f,0.f,0.f};
    v8f accq[4], acck[4], accv[4];
    #pragma unroll
    for (int n = 0; n < 4; ++n) { accq[n] = vzero; acck[n] = vzero; accv[n] = vzero; }

    const float* xrow = x + (size_t)(b*TT + t0 + lr)*CC;

    for (int kk = 0; kk < CC/32; ++kk) {
        // A-fragment 16x32: lane<16 holds K {0..7,16..23}, lane>=16 {8..15,24..31}
        const float* xp = xrow + kk*32 + hf*8;
        v4f a0 = *(const v4f*)(xp +  0);
        v4f a1 = *(const v4f*)(xp +  4);
        v4f a2 = *(const v4f*)(xp + 16);
        v4f a3 = *(const v4f*)(xp + 20);
        v16h af;
        #pragma unroll
        for (int e = 0; e < 4; ++e) {
            af[e]    = (_Float16)a0[e];
            af[4+e]  = (_Float16)a1[e];
            af[8+e]  = (_Float16)a2[e];
            af[12+e] = (_Float16)a3[e];
        }
        #pragma unroll
        for (int n = 0; n < 4; ++n) {
            v16h bq = load_w_bfrag(Whq, n, lr, hf, kk);
            accq[n] = __builtin_amdgcn_wmma_f32_16x16x32_f16(false, af, false, bq,
                        (short)0, accq[n], false, false);
            v16h bk = load_w_bfrag(Whk, n, lr, hf, kk);
            acck[n] = __builtin_amdgcn_wmma_f32_16x16x32_f16(false, af, false, bk,
                        (short)0, acck[n], false, false);
            v16h bv = load_w_bfrag(Whv, n, lr, hf, kk);
            accv[n] = __builtin_amdgcn_wmma_f32_16x16x32_f16(false, af, false, bv,
                        (short)0, accv[n], false, false);
        }
    }

    // Fused RoPE: the (h, h+32) pair lives in (tile n, tile n+2) of the SAME
    // lane, same accumulator slot -> pure per-lane register rotation.
    #pragma unroll
    for (int n = 0; n < 2; ++n) {
        const int h = n*16 + lr;                                   // 0..31
        const float invf = __powf(10000.0f, -(float)(2*h) * (1.0f/64.0f));
        #pragma unroll
        for (int r = 0; r < 8; ++r) {
            const float tpos = (float)(t0 + r + hf*8);
            const float ang  = tpos * invf;
            const float cs = __cosf(ang), sn = __sinf(ang);
            float q1 = accq[n][r], q2 = accq[n+2][r];
            accq[n][r]   = q1*cs - q2*sn;
            accq[n+2][r] = q2*cs + q1*sn;
            float k1 = acck[n][r], k2 = acck[n+2][r];
            acck[n][r]   = k1*cs - k2*sn;
            acck[n+2][r] = k2*cs + k1*sn;
        }
    }

    #pragma unroll
    for (int n = 0; n < 4; ++n) {
        const int h = n*16 + lr;
        #pragma unroll
        for (int r = 0; r < 8; ++r) {
            const int t = t0 + r + hf*8;
            const size_t o = (size_t)(b*TT + t)*HH + h;
            Qh[o] = (_Float16)accq[n][r];
            Kh[o] = (_Float16)acck[n][r];
        }
        v8h vv;
        #pragma unroll
        for (int r = 0; r < 8; ++r) vv[r] = (_Float16)accv[n][r];
        // V transposed [h][t]: lane holds fixed h, 8 consecutive t -> one b128 store
        *(v8h*)(Vt + ((size_t)b*HH + h)*TT + t0 + hf*8) = vv;
    }
}

// ---------------------------------------------------------------------------
// Kernel 2: causal flash attention, one wave per 16-row query tile,
// 32 keys per iteration: two 16-key S subtiles share one joint online-softmax
// update; P (16x32) goes through per-wave LDS once; P.V is a FULL K=32 WMMA.
// ---------------------------------------------------------------------------
__global__ __launch_bounds__(128) void flash_attn_kernel(
    const _Float16* __restrict__ Qh, const _Float16* __restrict__ Kh,
    const _Float16* __restrict__ Vt, float* __restrict__ out)
{
    __shared__ _Float16 lds_p[4][16*32];             // 1 KB per wave
    const int lane = threadIdx.x & 31;
    const int wave = threadIdx.x >> 5;
    const int tile = blockIdx.x * 4 + wave;
    const int b  = tile / NT;
    const int it = tile % NT;
    const int t0 = it * 16;
    const int hf = lane >> 4;
    const int lr = lane & 15;
    const float scale = 0.03125f;            // 1024^-0.5 (dim_model scaling)
    const float NEG   = -3.0e38f;            // finite -inf sentinel (no NaN paths)

    // Q A-fragments for both K-chunks of head dim (h 0..31, 32..63)
    v16h aq[2];
    {
        const _Float16* qrow = Qh + (size_t)(b*TT + t0 + lr)*HH;
        #pragma unroll
        for (int c = 0; c < 2; ++c) {
            v8h q0 = *(const v8h*)(qrow + c*32 + hf*8);
            v8h q1 = *(const v8h*)(qrow + c*32 + hf*8 + 16);
            #pragma unroll
            for (int e = 0; e < 8; ++e) { aq[c][e] = q0[e]; aq[c][8+e] = q1[e]; }
        }
    }

    const v8f vzero = {0.f,0.f,0.f,0.f,0.f,0.f,0.f,0.f};
    v8f acco[4];
    #pragma unroll
    for (int n = 0; n < 4; ++n) acco[n] = vzero;
    float m[8], l[8];
    #pragma unroll
    for (int r = 0; r < 8; ++r) { m[r] = NEG; l[r] = 0.0f; }

    const int nb = (it >> 1) + 1;            // 32-key blocks covering 0..t0+15(+pad)

    for (int jb = 0; jb < nb; ++jb) {
        const int s0 = jb * 32;

        // --- S = Q K^T for two 16-key subtiles ---------------------------
        v8f s[2];
        #pragma unroll
        for (int c2 = 0; c2 < 2; ++c2) {
            s[c2] = vzero;
            const _Float16* krow = Kh + (size_t)(b*TT + s0 + c2*16 + lr)*HH;
            #pragma unroll
            for (int c = 0; c < 2; ++c) {
                v8h k0 = *(const v8h*)(krow + c*32 + hf*8);
                v8h k1 = *(const v8h*)(krow + c*32 + hf*8 + 16);
                v16h bk;
                #pragma unroll
                for (int e = 0; e < 8; ++e) { bk[e] = k0[e]; bk[8+e] = k1[e]; }
                s[c2] = __builtin_amdgcn_wmma_f32_16x16x32_f16(false, aq[c], false, bk,
                            (short)0, s[c2], false, false);
            }
        }
        if (jb + 1 < nb)   // uniform branch: prefetch next K block
            __builtin_prefetch(Kh + (size_t)(b*TT + (jb+1)*32 + lr)*HH, 0, 0);

        // --- joint online softmax over 32 columns ------------------------
        const bool msk0 = (s0      + 15 > t0);       // uniform per wave
        const bool msk1 = (s0 + 16 + 15 > t0);
        float p0[8], p1[8];
        #pragma unroll
        for (int r = 0; r < 8; ++r) {
            const int M = r + hf*8;
            float sv0 = s[0][r] * scale;
            float sv1 = s[1][r] * scale;
            if (msk0 && (s0      + lr > t0 + M)) sv0 = NEG;
            if (msk1 && (s0 + 16 + lr > t0 + M)) sv1 = NEG;
            float rm = fmaxf(sv0, sv1);
            rm = fmaxf(rm, __shfl_xor(rm, 1));
            rm = fmaxf(rm, __shfl_xor(rm, 2));
            rm = fmaxf(rm, __shfl_xor(rm, 4));
            rm = fmaxf(rm, __shfl_xor(rm, 8));
            const float mnew  = fmaxf(m[r], rm);
            const float alpha = __expf(m[r] - mnew);
            float pe0 = __expf(sv0 - mnew);
            float pe1 = __expf(sv1 - mnew);
            float rs = pe0 + pe1;
            rs += __shfl_xor(rs, 1);
            rs += __shfl_xor(rs, 2);
            rs += __shfl_xor(rs, 4);
            rs += __shfl_xor(rs, 8);
            l[r] = l[r]*alpha + rs;
            m[r] = mnew;
            #pragma unroll
            for (int n = 0; n < 4; ++n) acco[n][r] *= alpha;
            p0[r] = pe0;
            p1[r] = pe1;
        }

        // --- redistribute P (16x32): D-layout -> A-layout via LDS --------
        #pragma unroll
        for (int r = 0; r < 8; ++r) {
            lds_p[wave][(r + hf*8)*32      + lr] = (_Float16)p0[r];
            lds_p[wave][(r + hf*8)*32 + 16 + lr] = (_Float16)p1[r];
        }
        asm volatile("s_wait_dscnt 0" ::: "memory");

        v16h ap;                     // full 16x32 A-fragment, no padding
        #pragma unroll
        for (int e = 0; e < 8; ++e) {
            ap[e]   = lds_p[wave][lr*32      + hf*8 + e];
            ap[8+e] = lds_p[wave][lr*32 + 16 + hf*8 + e];
        }

        // --- O += P . V, full K=32 (V^T gives contiguous b128 loads) -----
        #pragma unroll
        for (int n = 0; n < 4; ++n) {
            const _Float16* vp = Vt + ((size_t)b*HH + n*16 + lr)*TT + s0 + hf*8;
            v8h v0 = *(const v8h*)(vp);
            v8h v1 = *(const v8h*)(vp + 16);
            v16h bv;
            #pragma unroll
            for (int e = 0; e < 8; ++e) { bv[e] = v0[e]; bv[8+e] = v1[e]; }
            acco[n] = __builtin_amdgcn_wmma_f32_16x16x32_f16(false, ap, false, bv,
                        (short)0, acco[n], false, false);
        }
    }

    float linv[8];
    #pragma unroll
    for (int r = 0; r < 8; ++r) linv[r] = 1.0f / l[r];
    #pragma unroll
    for (int n = 0; n < 4; ++n) {
        #pragma unroll
        for (int r = 0; r < 8; ++r) {
            const int t = t0 + r + hf*8;
            out[(size_t)(b*TT + t)*HH + n*16 + lr] = acco[n][r] * linv[r];
        }
    }
}

// ---------------------------------------------------------------------------
extern "C" void kernel_launch(void* const* d_in, const int* in_sizes, int n_in,
                              void* d_out, int out_size, void* d_ws, size_t ws_size,
                              hipStream_t stream)
{
    (void)in_sizes; (void)n_in; (void)out_size;
    const float* x  = (const float*)d_in[0];
    const float* Wq = (const float*)d_in[1];
    const float* Wk = (const float*)d_in[2];
    const float* Wv = (const float*)d_in[3];
    float* out = (float*)d_out;

    const size_t mat  = (size_t)BB*TT*HH*sizeof(_Float16);   // 2 MB each
    const size_t wmat = (size_t)HH*CC*sizeof(_Float16);      // 128 KB each
    if (ws_size < 3*mat + 3*wmat) return;
    char* ws = (char*)d_ws;
    _Float16* Qh  = (_Float16*)(ws);
    _Float16* Kh  = (_Float16*)(ws + mat);
    _Float16* Vt  = (_Float16*)(ws + 2*mat);
    _Float16* Whq = (_Float16*)(ws + 3*mat);
    _Float16* Whk = (_Float16*)(ws + 3*mat + wmat);
    _Float16* Whv = (_Float16*)(ws + 3*mat + 2*wmat);

    wcvt_kernel<<<HH*CC/(256*4), 256, 0, stream>>>(Wq, Wk, Wv, Whq, Whk, Whv);
    dim3 grid(BB*TT/64), block(128);     // 4 waves/block, one 16-row tile per wave
    qkv_rope_kernel<<<grid, block, 0, stream>>>(x, Whq, Whk, Whv, Qh, Kh, Vt);
    flash_attn_kernel<<<grid, block, 0, stream>>>(Qh, Kh, Vt, out);
}
